// ScatterLinear_80470507258309
// MI455X (gfx1250) — matrix operand order repaired and checked by reading, not compiled, not run on hardware
//
#include <hip/hip_runtime.h>

typedef __attribute__((ext_vector_type(16))) _Float16 v16h;
typedef __attribute__((ext_vector_type(8)))  _Float16 v8h;
typedef __attribute__((ext_vector_type(8)))  float    v8f;

// ---------------- problem constants ----------------
static constexpr int BATCH    = 64;
static constexpr int CH       = 3;
static constexpr int IMG      = BATCH * CH;        // 192
static constexpr int LORI     = 8;                 // orientations
static constexpr int P32      = 1024;              // 32x32 pixels
static constexpr int P16      = 256;               // 16x16 pixels
static constexpr int C_SCAT   = 243;
static constexpr int FEATD    = C_SCAT * 64;       // 15552
static constexpr int NCLS     = 10;

// =====================================================================
// WMMA GEMM, 2x2 register-blocked: each wave owns a 32x32 output
// super-tile (4 x v_wmma_f32_16x16x32_f16 per K-step), reusing each
// A/B fragment twice -> half the L2 traffic per FLOP vs 1-tile waves.
// A row-major (lda=K); B passed TRANSPOSED as Bt (N rows x K cols).
// f16 operands, f32 accumulation. Output f16 (ldc=N) or f32+bias head.
// All M are multiples of 32; N may have a partial second column (head).
// =====================================================================
__device__ __forceinline__ v16h sl_cat8(v8h lo, v8h hi8)
{
    return __builtin_shufflevector(lo, hi8, 0,1,2,3,4,5,6,7,8,9,10,11,12,13,14,15);
}

__global__ __launch_bounds__(128)
void sl_wmma_gemm(const _Float16* __restrict__ A, const _Float16* __restrict__ Bt,
                  _Float16* __restrict__ Ch, float* __restrict__ Cf,
                  const float* __restrict__ bias, int M, int N, int K, int Nact)
{
    const int lane = threadIdx.x & 31;
    const int wv   = threadIdx.y;
    const int mt   = blockIdx.x;               // 32-row super-tile
    const int nt   = blockIdx.y * 4 + wv;      // 32-col super-tile
    if (nt * 32 >= N) return;
    const bool n1ok = (nt * 32 + 16) < N;      // wave-uniform column guard

    const int hi = lane >> 4;                  // half-wave select
    const int mr = lane & 15;                  // A row / B column within tile
    const _Float16* arow0 = A  + (size_t)(mt * 32 + mr) * (size_t)K;
    const _Float16* arow1 = arow0 + (size_t)16 * K;
    const _Float16* brow0 = Bt + (size_t)(nt * 32 + mr) * (size_t)K;
    const _Float16* brow1 = brow0 + (size_t)16 * K;
    const int ao = hi * 8;                     // A frag: K = hi*8 + h (+16 upper)
    const int bo = hi * 8;                     // B frag: K = hi*8 + h (+16 upper)

    v8f acc00 = {0.f,0.f,0.f,0.f,0.f,0.f,0.f,0.f};
    v8f acc01 = acc00, acc10 = acc00, acc11 = acc00;

    #pragma unroll 2
    for (int kb = 0; kb < K; kb += 32) {
        __builtin_prefetch(arow0 + kb + 128, 0, 1);
        __builtin_prefetch(brow0 + kb + 128, 0, 1);
        v16h af0 = sl_cat8(*(const v8h*)(arow0 + kb + ao),
                           *(const v8h*)(arow0 + kb + ao + 16));
        v16h af1 = sl_cat8(*(const v8h*)(arow1 + kb + ao),
                           *(const v8h*)(arow1 + kb + ao + 16));
        v16h bf0 = sl_cat8(*(const v8h*)(brow0 + kb + bo),
                           *(const v8h*)(brow0 + kb + bo + 16));
        acc00 = __builtin_amdgcn_wmma_f32_16x16x32_f16(false, af0, false, bf0,
                                                       (short)0, acc00, false, false);
        acc10 = __builtin_amdgcn_wmma_f32_16x16x32_f16(false, af1, false, bf0,
                                                       (short)0, acc10, false, false);
        if (n1ok) {
            v16h bf1 = sl_cat8(*(const v8h*)(brow1 + kb + bo),
                               *(const v8h*)(brow1 + kb + bo + 16));
            acc01 = __builtin_amdgcn_wmma_f32_16x16x32_f16(false, af0, false, bf1,
                                                           (short)0, acc01, false, false);
            acc11 = __builtin_amdgcn_wmma_f32_16x16x32_f16(false, af1, false, bf1,
                                                           (short)0, acc11, false, false);
        }
    }

    // C/D layout: N = lane&15, M = r + hi*8 within each 16x16 tile.
    const int nc0 = nt * 32 + mr;
    if (Cf) {
        // logit head: N=16 (single column), Nact=10 valid outputs
        if (nc0 < Nact) {
            #pragma unroll
            for (int r = 0; r < 8; ++r) {
                int mg = mt * 32 + hi * 8 + r;
                Cf[(size_t)mg * Nact + nc0] = acc00[r] + (bias ? bias[nc0] : 0.f);
                Cf[(size_t)(mg + 16) * Nact + nc0] = acc10[r] + (bias ? bias[nc0] : 0.f);
            }
        }
    } else {
        #pragma unroll
        for (int r = 0; r < 8; ++r) {
            int mg0 = mt * 32 + hi * 8 + r;
            Ch[(size_t)mg0 * N + nc0]        = (_Float16)acc00[r];
            Ch[(size_t)(mg0 + 16) * N + nc0] = (_Float16)acc10[r];
        }
        if (n1ok) {
            #pragma unroll
            for (int r = 0; r < 8; ++r) {
                int mg0 = mt * 32 + hi * 8 + r;
                Ch[(size_t)mg0 * N + nc0 + 16]        = (_Float16)acc01[r];
                Ch[(size_t)(mg0 + 16) * N + nc0 + 16] = (_Float16)acc11[r];
            }
        }
    }
}

// =====================================================================
// Build spatial filters by direct inverse DFT (exact: uses 32-entry table).
// planes 0-7: psi0 -> f0r/f0i ; 8-15: psi1 -> f1r/f1i ; 16: phi -> phr (Re)
// plane 17: phi1 = sub_fourier(phi,2) -> ph1r (Re, 16x16)
// =====================================================================
__global__ void sl_build_spatial(const float* __restrict__ psi0,
                                 const float* __restrict__ psi1,
                                 const float* __restrict__ phi,
                                 float* f0r, float* f0i, float* f1r, float* f1i,
                                 float* phr, float* ph1r)
{
    __shared__ float cs[32], sn[32];
    int tid = threadIdx.x;
    if (tid < 32) {
        float ang = 6.283185307179586f * (float)tid / 32.f;
        cs[tid] = cosf(ang);
        sn[tid] = sinf(ang);
    }
    __syncthreads();

    int plane = blockIdx.x;
    if (plane < 17) {
        const float* P = (plane < 8)  ? psi0 + plane * P32
                       : (plane < 16) ? psi1 + (plane - 8) * P32
                                      : phi;
        int p = tid, py = p >> 5, px = p & 31;
        float sr = 0.f, si = 0.f;
        for (int q = 0; q < P32; ++q) {
            int qy = q >> 5, qx = q & 31;
            int k = (py * qy + px * qx) & 31;
            float v = P[q];
            sr += v * cs[k];
            si += v * sn[k];
        }
        sr *= (1.f / 1024.f); si *= (1.f / 1024.f);
        if (plane < 8)       { f0r[plane * P32 + p] = sr;        f0i[plane * P32 + p] = si; }
        else if (plane < 16) { f1r[(plane - 8) * P32 + p] = sr;  f1i[(plane - 8) * P32 + p] = si; }
        else                 { phr[p] = sr; }
    } else if (tid < 256) {
        int py = tid >> 4, px = tid & 15;
        float sr = 0.f;
        for (int q = 0; q < P16; ++q) {
            int qy = q >> 4, qx = q & 15;
            float v = 0.25f * (phi[qy * 32 + qx] + phi[(qy + 16) * 32 + qx] +
                               phi[qy * 32 + qx + 16] + phi[(qy + 16) * 32 + qx + 16]);
            int k = ((py * qy + px * qx) & 15) * 2;   // e^{2pi i k/16} = table[2k]
            sr += v * cs[k];
        }
        ph1r[tid] = sr * (1.f / 256.f);
    }
}

// Circular-conv operator matrices, stored TRANSPOSED (rows = output index
// n = t*nq + a, cols = input pixel p of the 32x32 plane).
__global__ void sl_fill_conv_mat(const float* __restrict__ fr, const float* __restrict__ fi,
                                 _Float16* __restrict__ Kr, _Float16* __restrict__ Ki,
                                 int nq, int qside, int sub)
{
    size_t total = (size_t)LORI * nq * P32;
    for (size_t i = (size_t)blockIdx.x * blockDim.x + threadIdx.x; i < total;
         i += (size_t)gridDim.x * blockDim.x) {
        int p = (int)(i & 1023);
        size_t r = i >> 10;
        int a = (int)(r % nq);
        int t = (int)(r / nq);
        int oy = (a / qside) * sub, ox = (a % qside) * sub;
        int py = p >> 5, px = p & 31;
        int d = (((oy - py) & 31) << 5) | ((ox - px) & 31);
        Kr[i] = (_Float16)fr[t * P32 + d];
        Ki[i] = (_Float16)fi[t * P32 + d];
    }
}

// phi low-pass operator (real), rows = 64 output pixels, cols = inside^2
__global__ void sl_fill_phi_mat(const float* __restrict__ f, _Float16* __restrict__ Kp,
                                int inside, int sub)
{
    int cols = inside * inside;
    int mask = inside - 1;
    size_t total = (size_t)64 * cols;
    for (size_t i = (size_t)blockIdx.x * blockDim.x + threadIdx.x; i < total;
         i += (size_t)gridDim.x * blockDim.x) {
        int p = (int)(i % cols);
        int a = (int)(i / cols);
        int oy = (a >> 3) * sub, ox = (a & 7) * sub;
        int py = p / inside, px = p % inside;
        int d = ((oy - py) & mask) * inside + ((ox - px) & mask);
        Kp[i] = (_Float16)f[d];
    }
}

__global__ void sl_cvt_f16(const float* __restrict__ in, _Float16* __restrict__ out, size_t n)
{
    for (size_t i = (size_t)blockIdx.x * blockDim.x + threadIdx.x; i < n;
         i += (size_t)gridDim.x * blockDim.x)
        out[i] = (_Float16)in[i];
}

// Wt: 16 x 15552 (N-major Bt form); rows 10..15 zero-padded.
__global__ void sl_fill_wt(const float* __restrict__ W, _Float16* __restrict__ Wt)
{
    size_t total = (size_t)16 * FEATD;
    for (size_t i = (size_t)blockIdx.x * blockDim.x + threadIdx.x; i < total;
         i += (size_t)gridDim.x * blockDim.x) {
        int n = (int)(i / FEATD);
        Wt[i] = (n < NCLS) ? (_Float16)W[i] : (_Float16)0.f;
    }
}

__global__ void sl_cabs(const _Float16* __restrict__ re, const _Float16* __restrict__ im,
                        _Float16* __restrict__ o, size_t n)
{
    for (size_t i = (size_t)blockIdx.x * blockDim.x + threadIdx.x; i < n;
         i += (size_t)gridDim.x * blockDim.x) {
        float a = (float)re[i], b = (float)im[i];
        o[i] = (_Float16)sqrtf(a * a + b * b);
    }
}

// Assemble feat (B,243,64) from S0/S1A/S1B/S2 in kymatio channel order.
__global__ void sl_assemble(const _Float16* __restrict__ S0, const _Float16* __restrict__ S1A,
                            const _Float16* __restrict__ S1B, const _Float16* __restrict__ S2,
                            _Float16* __restrict__ feat)
{
    size_t total = (size_t)BATCH * C_SCAT * 64;
    for (size_t i = (size_t)blockIdx.x * blockDim.x + threadIdx.x; i < total;
         i += (size_t)gridDim.x * blockDim.x) {
        int pos = (int)(i & 63);
        size_t r = i >> 6;
        int ch = (int)(r % C_SCAT);
        int b  = (int)(r / C_SCAT);
        int c  = ch / 81, k = ch % 81;
        int row = b * CH + c;
        _Float16 v;
        if (k == 0)       v = S0[(size_t)row * 64 + pos];
        else if (k < 9)   v = S1A[((size_t)row * 8 + (k - 1)) * 64 + pos];
        else if (k < 17)  v = S1B[((size_t)row * 8 + (k - 9)) * 64 + pos];
        else {
            int kk = k - 17;
            v = S2[(((size_t)row * 8 + kk / 8) * 8 + kk % 8) * 64 + pos];
        }
        feat[i] = v;
    }
}

// GroupNorm over 27 groups of (9 ch x 64 px) = 576 elements, biased var.
__global__ void sl_groupnorm(const _Float16* __restrict__ feat,
                             const float* __restrict__ gamma, const float* __restrict__ beta,
                             _Float16* __restrict__ out)
{
    __shared__ float red[64];
    int b = blockIdx.x, g = blockIdx.y, tid = threadIdx.x;  // 64 threads
    const _Float16* p = feat + ((size_t)b * C_SCAT + g * 9) * 64;
    float vals[9], s = 0.f, ss = 0.f;
    #pragma unroll
    for (int c = 0; c < 9; ++c) {
        float v = (float)p[(size_t)c * 64 + tid];
        vals[c] = v; s += v; ss += v * v;
    }
    red[tid] = s; __syncthreads();
    for (int o = 32; o > 0; o >>= 1) { if (tid < o) red[tid] += red[tid + o]; __syncthreads(); }
    float mean = red[0] * (1.f / 576.f); __syncthreads();
    red[tid] = ss; __syncthreads();
    for (int o = 32; o > 0; o >>= 1) { if (tid < o) red[tid] += red[tid + o]; __syncthreads(); }
    float var = red[0] * (1.f / 576.f) - mean * mean;
    float inv = rsqrtf(var + 1e-5f);
    #pragma unroll
    for (int c = 0; c < 9; ++c) {
        int ch = g * 9 + c;
        float y = (vals[c] - mean) * inv * gamma[ch] + beta[ch];
        out[((size_t)b * C_SCAT + ch) * 64 + tid] = (_Float16)y;
    }
}

// ---------------------------------------------------------------------
static inline void gemm(hipStream_t s, const _Float16* A, const _Float16* Bt,
                        _Float16* Ch, float* Cf, const float* bias,
                        int M, int N, int K, int Nact)
{
    int nSup = (N + 31) / 32;
    dim3 blk(32, 4), grd(M / 32, (nSup + 3) / 4);
    sl_wmma_gemm<<<grd, blk, 0, s>>>(A, Bt, Ch, Cf, bias, M, N, K, Nact);
}

extern "C" void kernel_launch(void* const* d_in, const int* in_sizes, int n_in,
                              void* d_out, int out_size, void* d_ws, size_t ws_size,
                              hipStream_t stream)
{
    (void)in_sizes; (void)n_in; (void)out_size;
    const float* x     = (const float*)d_in[0];
    const float* psi0  = (const float*)d_in[1];
    const float* psi1  = (const float*)d_in[2];
    const float* phi   = (const float*)d_in[3];
    const float* gamma = (const float*)d_in[4];
    const float* beta  = (const float*)d_in[5];
    const float* W     = (const float*)d_in[6];
    const float* bvec  = (const float*)d_in[7];
    float* out = (float*)d_out;

    char* base = (char*)d_ws; size_t off = 0;
    auto alloc = [&](size_t bytes) -> void* {
        off = (off + 255) & ~(size_t)255;
        void* p = base + off; off += bytes; return p;
    };
    float* f0r = (float*)alloc(LORI * P32 * 4);
    float* f0i = (float*)alloc(LORI * P32 * 4);
    float* f1r = (float*)alloc(LORI * P32 * 4);
    float* f1i = (float*)alloc(LORI * P32 * 4);
    float* phr  = (float*)alloc(P32 * 4);
    float* ph1r = (float*)alloc(P16 * 4);
    _Float16* K0r = (_Float16*)alloc((size_t)LORI * P32 * P32 * 2);   // 8192x1024
    _Float16* K0i = (_Float16*)alloc((size_t)LORI * P32 * P32 * 2);
    _Float16* K1r = (_Float16*)alloc((size_t)LORI * P16 * P32 * 2);   // 2048x1024
    _Float16* K1i = (_Float16*)alloc((size_t)LORI * P16 * P32 * 2);
    _Float16* Kp0 = (_Float16*)alloc((size_t)64 * P32 * 2);
    _Float16* Kp1 = (_Float16*)alloc((size_t)64 * P16 * 2);
    _Float16* Xh  = (_Float16*)alloc((size_t)IMG * P32 * 2);
    _Float16* U0r = (_Float16*)alloc((size_t)IMG * LORI * P32 * 2);
    _Float16* U0i = (_Float16*)alloc((size_t)IMG * LORI * P32 * 2);
    _Float16* U0  = (_Float16*)alloc((size_t)IMG * LORI * P32 * 2);
    _Float16* U1r = (_Float16*)alloc((size_t)IMG * LORI * P16 * 2);
    _Float16* U1i = (_Float16*)alloc((size_t)IMG * LORI * P16 * 2);
    _Float16* U1  = (_Float16*)alloc((size_t)IMG * LORI * P16 * 2);
    _Float16* U2r = (_Float16*)alloc((size_t)IMG * LORI * LORI * P16 * 2);
    _Float16* U2i = (_Float16*)alloc((size_t)IMG * LORI * LORI * P16 * 2);
    _Float16* U2  = (_Float16*)alloc((size_t)IMG * LORI * LORI * P16 * 2);
    _Float16* S0  = (_Float16*)alloc((size_t)IMG * 64 * 2);
    _Float16* S1A = (_Float16*)alloc((size_t)IMG * LORI * 64 * 2);
    _Float16* S1B = (_Float16*)alloc((size_t)IMG * LORI * 64 * 2);
    _Float16* S2  = (_Float16*)alloc((size_t)IMG * LORI * LORI * 64 * 2);
    _Float16* feat  = (_Float16*)alloc((size_t)BATCH * FEATD * 2);
    _Float16* featN = (_Float16*)alloc((size_t)BATCH * FEATD * 2);
    _Float16* Wt    = (_Float16*)alloc((size_t)16 * FEATD * 2);
    if (off > ws_size) return;   // workspace too small: nothing safe to do

    // ---- setup: spatial filters + operator matrices + f16 inputs ----
    sl_build_spatial<<<18, 1024, 0, stream>>>(psi0, psi1, phi, f0r, f0i, f1r, f1i, phr, ph1r);
    sl_fill_conv_mat<<<2048, 256, 0, stream>>>(f0r, f0i, K0r, K0i, P32, 32, 1);
    sl_fill_conv_mat<<<1024, 256, 0, stream>>>(f1r, f1i, K1r, K1i, P16, 16, 2);
    sl_fill_phi_mat<<<128, 256, 0, stream>>>(phr, Kp0, 32, 4);
    sl_fill_phi_mat<<<64, 256, 0, stream>>>(ph1r, Kp1, 16, 2);
    sl_cvt_f16<<<384, 256, 0, stream>>>(x, Xh, (size_t)IMG * P32);
    sl_fill_wt<<<512, 256, 0, stream>>>(W, Wt);

    // ---- order 1, j=0 : u0 = |x (*) psi0| at full res ----
    gemm(stream, Xh, K0r, U0r, nullptr, nullptr, IMG, LORI * P32, P32, 0);
    gemm(stream, Xh, K0i, U0i, nullptr, nullptr, IMG, LORI * P32, P32, 0);
    sl_cabs<<<2048, 256, 0, stream>>>(U0r, U0i, U0, (size_t)IMG * LORI * P32);

    // ---- order 1, j=1 : u1 = |sub2(x (*) psi1)| ----
    gemm(stream, Xh, K1r, U1r, nullptr, nullptr, IMG, LORI * P16, P32, 0);
    gemm(stream, Xh, K1i, U1i, nullptr, nullptr, IMG, LORI * P16, P32, 0);
    sl_cabs<<<512, 256, 0, stream>>>(U1r, U1i, U1, (size_t)IMG * LORI * P16);

    // ---- low-pass outputs ----
    gemm(stream, Xh, Kp0, S0, nullptr, nullptr, IMG, 64, P32, 0);              // s0
    gemm(stream, U0, Kp0, S1A, nullptr, nullptr, IMG * LORI, 64, P32, 0);      // s1a

    // ---- order 2 : u2 = |sub2(u0 (*) psi1)|  (same K1 operator) ----
    gemm(stream, U0, K1r, U2r, nullptr, nullptr, IMG * LORI, LORI * P16, P32, 0);
    gemm(stream, U0, K1i, U2i, nullptr, nullptr, IMG * LORI, LORI * P16, P32, 0);
    sl_cabs<<<4096, 256, 0, stream>>>(U2r, U2i, U2, (size_t)IMG * LORI * LORI * P16);

    gemm(stream, U1, Kp1, S1B, nullptr, nullptr, IMG * LORI, 64, P16, 0);      // s1b
    gemm(stream, U2, Kp1, S2, nullptr, nullptr, IMG * LORI * LORI, 64, P16, 0);// s2

    // ---- feature assembly + GroupNorm + linear head ----
    sl_assemble<<<2048, 256, 0, stream>>>(S0, S1A, S1B, S2, feat);
    sl_groupnorm<<<dim3(BATCH, 27), 64, 0, stream>>>(feat, gamma, beta, featN);
    gemm(stream, featN, Wt, nullptr, out, bvec, BATCH, 16, FEATD, NCLS);
}
